// DGCNN_80917183857081
// MI455X (gfx1250) — matrix operand order repaired and checked by reading, not compiled, not run on hardware
//
#include <hip/hip_runtime.h>
#include <hip/hip_bf16.h>

#define BATCH 16
#define NPTS  2048
#define KNN   20
#define LB    (NPTS*KNN)   /* 40960 columns per batch */
#define BNEPS 1e-5f

typedef __bf16 v16bf __attribute__((ext_vector_type(16)));
typedef float  v8f   __attribute__((ext_vector_type(8)));

struct __attribute__((aligned(16))) U128 { unsigned int a0, a1, a2, a3; };
union FragBF { v16bf v; U128 q[2]; };

__device__ __forceinline__ float leaky_f(float v) { return v > 0.f ? v : 0.2f * v; }

// -------- squared norms: xx[b*N+n] = sum_c x[b,c,n]^2 --------
__global__ void xx_kernel(const float* __restrict__ x, float* __restrict__ xx, int C) {
    int i = blockIdx.x * blockDim.x + threadIdx.x;
    if (i >= BATCH * NPTS) return;
    int b = i / NPTS, n = i - b * NPTS;
    const float* xb = x + (size_t)b * C * NPTS + n;
    float s = 0.f;
    for (int c = 0; c < C; ++c) { float v = xb[(size_t)c * NPTS]; s += v * v; }
    xx[i] = s;
}

// -------- kNN top-20 on pd = 2*x_n.x_m - ||x_n||^2 - ||x_m||^2 --------
// One block = 256 rows of one batch. LDS: row block 256 x (C+1), column tile C x 128.
__global__ void knn_kernel(const float* __restrict__ x, const float* __restrict__ xx,
                           int* __restrict__ idx, int C) {
    extern __shared__ float smem[];
    const int tid = threadIdx.x;
    const int b = blockIdx.y;
    const int nbase = blockIdx.x * 256;
    const int n = nbase + tid;
    const int stride = C + 1;                 // pad to avoid 64-bank conflicts
    float* xrow = smem;                       // 256 * (C+1)
    float* xcol = smem + 256 * stride;        // C * 128
    const float* xb = x + (size_t)b * C * NPTS;

    for (int i = tid; i < 256 * C; i += 256) {
        int r = i / C, c = i - r * C;
        xrow[r * stride + c] = xb[(size_t)c * NPTS + nbase + r];
    }
    __syncthreads();

    float vals[KNN]; int ids[KNN];
#pragma unroll
    for (int j = 0; j < KNN; ++j) { vals[j] = -3.0e38f; ids[j] = 0; }

    const float xxn = xx[b * NPTS + n];
    const float* xrn = xrow + tid * stride;

    for (int mbase = 0; mbase < NPTS; mbase += 128) {
        __syncthreads();
        for (int i = tid; i < C * 128; i += 256) {
            int c = i >> 7, m = i & 127;
            xcol[c * 128 + m] = xb[(size_t)c * NPTS + mbase + m];
        }
        __syncthreads();
        for (int mm = 0; mm < 128; ++mm) {
            float dot = 0.f;
            for (int c = 0; c < C; ++c) dot += xrn[c] * xcol[c * 128 + mm];
            float pd = 2.f * dot - xxn - xx[b * NPTS + mbase + mm];
            if (pd > vals[KNN - 1]) {
                // branchless sorted insert, static indices only (stays in VGPRs)
#pragma unroll
                for (int j = KNN - 1; j > 0; --j) {
                    bool gp = pd > vals[j - 1];
                    bool gj = pd > vals[j];
                    vals[j] = gp ? vals[j - 1] : (gj ? pd : vals[j]);
                    ids[j]  = gp ? ids[j - 1]  : (gj ? (mbase + mm) : ids[j]);
                }
                if (pd > vals[0]) { vals[0] = pd; ids[0] = mbase + mm; }
            }
        }
    }
    int* op = idx + ((size_t)(b * NPTS) + n) * KNN;
#pragma unroll
    for (int j = 0; j < KNN; ++j) op[j] = ids[j];
}

// -------- weights fp32 -> bf16, K-padded to C2p --------
__global__ void wcvt_kernel(const float* __restrict__ W, __hip_bfloat16* __restrict__ Wbf,
                            int O, int C2, int C2p) {
    int i = blockIdx.x * blockDim.x + threadIdx.x;
    if (i >= O * C2p) return;
    int o = i / C2p, c = i - o * C2p;
    float v = (c < C2) ? W[o * C2 + c] : 0.f;
    Wbf[i] = __float2bfloat16(v);
}

// -------- edge features for one batch, bf16 [LB][C2p]: [x_j-x_n | x_n | pad] --------
__global__ void feat_kernel(const float* __restrict__ x, const int* __restrict__ idx,
                            __hip_bfloat16* __restrict__ F, int b, int cin, int C2p) {
    int nk = blockIdx.x * blockDim.x + threadIdx.x;
    if (nk >= LB) return;
    int n = nk / KNN;
    int j = idx[((size_t)(b * NPTS) + n) * KNN + (nk - n * KNN)];
    const float* xb = x + (size_t)b * cin * NPTS;
    __hip_bfloat16* Fo = F + (size_t)nk * C2p;
    for (int c = 0; c < cin; ++c) {
        float xn = xb[(size_t)c * NPTS + n];
        float xj = xb[(size_t)c * NPTS + j];
        Fo[c]       = __float2bfloat16(xj - xn);
        Fo[cin + c] = __float2bfloat16(xn);
    }
    for (int c = 2 * cin; c < C2p; ++c) Fo[c] = __float2bfloat16(0.f);
}

// -------- WMMA GEMM: Y[o][l] = sum_c A[o][c] * Bm[l][c], bf16 in / f32 out --------
// Register-blocked: each wave owns a 32x32 macro-tile = 2x2 WMMA tiles, so every
// A/B fragment pair feeds two v_wmma_f32_16x16x32_bf16 ops (4 WMMA per 8 b128 loads).
// Fragment layouts per CDNA5 ISA 7.12.2 (16-bit A 16x32, 16-bit B 32x16, f32 C/D 16x16).
__global__ void __launch_bounds__(256) wmma_gemm_kernel(
    const __hip_bfloat16* __restrict__ A,   // [O][C2p] row-major bf16
    const __hip_bfloat16* __restrict__ Bm,  // [L][C2p] row-major bf16
    float* __restrict__ Y,                  // [O][L] f32
    int O, int C2p, int L) {
    const int lane = threadIdx.x & 31;
    const int wave = threadIdx.x >> 5;          // 8 wave32 waves
    const int wm = wave >> 2, wl = wave & 3;    // 2 (O) x 4 (L) wave grid
    const int otile = blockIdx.y * 64 + wm * 32;   // wave covers O rows [otile, otile+32)
    const int ltile = blockIdx.x * 128 + wl * 32;  // wave covers L cols [ltile, ltile+32)
    const int h = lane >> 4, q = lane & 15;

    const char* arow0 = (const char*)(A + (size_t)(otile + q) * C2p);
    const char* arow1 = (const char*)(A + (size_t)(otile + 16 + q) * C2p);
    const char* brow0 = (const char*)(Bm + (size_t)(ltile + q) * C2p);
    const char* brow1 = (const char*)(Bm + (size_t)(ltile + 16 + q) * C2p);

    v8f acc00 = {}, acc01 = {}, acc10 = {}, acc11 = {};
    for (int kk = 0; kk < C2p; kk += 32) {
        FragBF a0, a1, b0, b1;
        // A lane m(+16h): row m, K = kk+8h..+7 and kk+16+8h..+7
        a0.q[0] = *(const U128*)(arow0 + 2 * (kk + 8 * h));
        a0.q[1] = *(const U128*)(arow0 + 2 * (kk + 16 + 8 * h));
        a1.q[0] = *(const U128*)(arow1 + 2 * (kk + 8 * h));
        a1.q[1] = *(const U128*)(arow1 + 2 * (kk + 16 + 8 * h));
        // B lane n(+16h): col n, K = kk+8h..+7 and kk+16+8h..+7
        b0.q[0] = *(const U128*)(brow0 + 2 * (kk + 8 * h));
        b0.q[1] = *(const U128*)(brow0 + 2 * (kk + 16 + 8 * h));
        b1.q[0] = *(const U128*)(brow1 + 2 * (kk + 8 * h));
        b1.q[1] = *(const U128*)(brow1 + 2 * (kk + 16 + 8 * h));
        // prefetch next K-step of the streaming B rows (global_prefetch_b8)
        __builtin_prefetch(brow0 + 2 * (kk + 32 + 16 * h), 0, 1);
        __builtin_prefetch(brow1 + 2 * (kk + 32 + 16 * h), 0, 1);
        acc00 = __builtin_amdgcn_wmma_f32_16x16x32_bf16(false, a0.v, false, b0.v,
                                                        (short)0, acc00, false, false);
        acc01 = __builtin_amdgcn_wmma_f32_16x16x32_bf16(false, a0.v, false, b1.v,
                                                        (short)0, acc01, false, false);
        acc10 = __builtin_amdgcn_wmma_f32_16x16x32_bf16(false, a1.v, false, b0.v,
                                                        (short)0, acc10, false, false);
        acc11 = __builtin_amdgcn_wmma_f32_16x16x32_bf16(false, a1.v, false, b1.v,
                                                        (short)0, acc11, false, false);
    }
    // C/D: lanes 0-15 -> M=r, lanes 16-31 -> M=8+r, N = q (per 16x16 tile)
    float* y00 = Y + (size_t)(otile + 8 * h) * L + ltile + q;
    float* y01 = y00 + 16;
    float* y10 = Y + (size_t)(otile + 16 + 8 * h) * L + ltile + q;
    float* y11 = y10 + 16;
#pragma unroll
    for (int r = 0; r < 8; ++r) {
        y00[(size_t)r * L] = acc00[r];
        y01[(size_t)r * L] = acc01[r];
        y10[(size_t)r * L] = acc10[r];
        y11[(size_t)r * L] = acc11[r];
    }
}

// -------- epilogue: max over k, accumulate per-channel sum / sumsq --------
__global__ void epilogue_kernel(const float* __restrict__ Yb, float* __restrict__ ymax,
                                float* __restrict__ stats, int b, int O) {
    __shared__ float ssum[256], ssq[256];
    int o = blockIdx.y;
    int n = blockIdx.x * 256 + threadIdx.x;
    const float* yp = Yb + (size_t)o * LB + n * KNN;
    float mx = -3.0e38f, s = 0.f, sq = 0.f;
#pragma unroll
    for (int k = 0; k < KNN; ++k) { float v = yp[k]; mx = fmaxf(mx, v); s += v; sq += v * v; }
    ymax[((size_t)b * O + o) * NPTS + n] = mx;
    ssum[threadIdx.x] = s; ssq[threadIdx.x] = sq;
    __syncthreads();
    for (int w = 128; w > 0; w >>= 1) {
        if (threadIdx.x < w) { ssum[threadIdx.x] += ssum[threadIdx.x + w]; ssq[threadIdx.x] += ssq[threadIdx.x + w]; }
        __syncthreads();
    }
    if (threadIdx.x == 0) {
        atomicAdd(&stats[o], ssum[0]);
        atomicAdd(&stats[1024 + o], ssq[0]);
    }
}

__global__ void zero_kernel(float* p, int n) {
    int i = blockIdx.x * blockDim.x + threadIdx.x;
    if (i < n) p[i] = 0.f;
}

// -------- BN(+leaky) on k-maxed tensor (g>0 => commutes with max_k) --------
__global__ void apply_bn_kernel(const float* __restrict__ ymax, const float* __restrict__ stats,
                                const float* __restrict__ g, const float* __restrict__ beta,
                                float* __restrict__ xo, int O) {
    int i = blockIdx.x * blockDim.x + threadIdx.x;
    if (i >= BATCH * O * NPTS) return;
    int o = (i / NPTS) % O;
    const float cnt = (float)(BATCH * NPTS * KNN);
    float m = stats[o] / cnt;
    float var = stats[1024 + o] / cnt - m * m;
    float rstd = rsqrtf(var + BNEPS);
    float v = (ymax[i] - m) * rstd * g[o] + beta[o];
    xo[i] = leaky_f(v);
}

// -------- pack h = concat(x1..x4) as bf16 [B*N][512] --------
__global__ void pack_h_kernel(const float* __restrict__ x1, const float* __restrict__ x2,
                              const float* __restrict__ x3, const float* __restrict__ x4,
                              __hip_bfloat16* __restrict__ h) {
    int l = blockIdx.x * blockDim.x + threadIdx.x;   // b*N+n
    if (l >= BATCH * NPTS) return;
    int b = l / NPTS, n = l - b * NPTS;
    __hip_bfloat16* hp = h + (size_t)l * 512;
    for (int c = 0; c < 64; ++c)  hp[c]       = __float2bfloat16(x1[((size_t)b * 64 + c) * NPTS + n]);
    for (int c = 0; c < 64; ++c)  hp[64 + c]  = __float2bfloat16(x2[((size_t)b * 64 + c) * NPTS + n]);
    for (int c = 0; c < 128; ++c) hp[128 + c] = __float2bfloat16(x3[((size_t)b * 128 + c) * NPTS + n]);
    for (int c = 0; c < 256; ++c) hp[256 + c] = __float2bfloat16(x4[((size_t)b * 256 + c) * NPTS + n]);
}

// -------- per-channel mean/rstd of Y5 over all b,n --------
__global__ void stats5_kernel(const float* __restrict__ Y5, float* __restrict__ mean,
                              float* __restrict__ rstd) {
    __shared__ float ssum[256], ssq[256];
    int o = blockIdx.x;
    const float* yp = Y5 + (size_t)o * (BATCH * NPTS);
    float s = 0.f, sq = 0.f;
    for (int i = threadIdx.x; i < BATCH * NPTS; i += 256) { float v = yp[i]; s += v; sq += v * v; }
    ssum[threadIdx.x] = s; ssq[threadIdx.x] = sq;
    __syncthreads();
    for (int w = 128; w > 0; w >>= 1) {
        if (threadIdx.x < w) { ssum[threadIdx.x] += ssum[threadIdx.x + w]; ssq[threadIdx.x] += ssq[threadIdx.x + w]; }
        __syncthreads();
    }
    if (threadIdx.x == 0) {
        float m = ssum[0] / (float)(BATCH * NPTS);
        mean[o] = m;
        rstd[o] = rsqrtf(ssq[0] / (float)(BATCH * NPTS) - m * m + BNEPS);
    }
}

// -------- p[b] = [max_n y5, mean_n y5] after bn+leaky --------
__global__ void pool_kernel(const float* __restrict__ Y5, const float* __restrict__ mean,
                            const float* __restrict__ rstd, const float* __restrict__ g,
                            const float* __restrict__ beta, float* __restrict__ p) {
    int i = blockIdx.x * blockDim.x + threadIdx.x;   // b*1024+o
    if (i >= BATCH * 1024) return;
    int b = i / 1024, o = i - b * 1024;
    const float* yp = Y5 + (size_t)o * (BATCH * NPTS) + b * NPTS;
    float m = mean[o], rs = rstd[o], gg = g[o], bb = beta[o];
    float mx = -3.0e38f, s = 0.f;
    for (int n = 0; n < NPTS; ++n) {
        float v = leaky_f((yp[n] - m) * rs * gg + bb);
        mx = fmaxf(mx, v); s += v;
    }
    p[b * 2048 + o]        = mx;
    p[b * 2048 + 1024 + o] = s / (float)NPTS;
}

// -------- z = p @ Wl^T  (16x256, tiny) --------
__global__ void fc_kernel(const float* __restrict__ p, const float* __restrict__ Wl,
                          float* __restrict__ z) {
    int i = blockIdx.x * blockDim.x + threadIdx.x;   // b*256+o
    if (i >= BATCH * 256) return;
    int b = i / 256, o = i - b * 256;
    const float* pp = p + b * 2048;
    const float* wp = Wl + (size_t)o * 2048;
    float s = 0.f;
    for (int c = 0; c < 2048; ++c) s += pp[c] * wp[c];
    z[i] = s;
}

// -------- batch-norm over batch axis, write output --------
__global__ void final_bn_kernel(const float* __restrict__ z, const float* __restrict__ g,
                                const float* __restrict__ beta, float* __restrict__ out) {
    int o = blockIdx.x * blockDim.x + threadIdx.x;
    if (o >= 256) return;
    float s = 0.f;
    for (int b = 0; b < BATCH; ++b) s += z[b * 256 + o];
    float m = s / (float)BATCH;
    float v = 0.f;
    for (int b = 0; b < BATCH; ++b) { float d = z[b * 256 + o] - m; v += d * d; }
    v /= (float)BATCH;
    float rs = rsqrtf(v + BNEPS);
    for (int b = 0; b < BATCH; ++b) out[b * 256 + o] = (z[b * 256 + o] - m) * rs * g[o] + beta[o];
}

extern "C" void kernel_launch(void* const* d_in, const int* in_sizes, int n_in,
                              void* d_out, int out_size, void* d_ws, size_t ws_size,
                              hipStream_t stream) {
    const float* x  = (const float*)d_in[0];
    const float* W1 = (const float*)d_in[1];  const float* g1 = (const float*)d_in[2];  const float* b1 = (const float*)d_in[3];
    const float* W2 = (const float*)d_in[4];  const float* g2 = (const float*)d_in[5];  const float* b2 = (const float*)d_in[6];
    const float* W3 = (const float*)d_in[7];  const float* g3 = (const float*)d_in[8];  const float* b3 = (const float*)d_in[9];
    const float* W4 = (const float*)d_in[10]; const float* g4 = (const float*)d_in[11]; const float* b4 = (const float*)d_in[12];
    const float* W5 = (const float*)d_in[13]; const float* g5 = (const float*)d_in[14]; const float* b5 = (const float*)d_in[15];
    const float* Wl = (const float*)d_in[16]; const float* g6 = (const float*)d_in[17]; const float* b6 = (const float*)d_in[18];
    float* out = (float*)d_out;
    (void)in_sizes; (void)n_in; (void)out_size; (void)ws_size;

    char* ws = (char*)d_ws;
    size_t off = 0;
    auto walloc = [&](size_t bytes) -> char* {
        char* p = ws + off; off += (bytes + 255) & ~(size_t)255; return p;
    };
    float* xx    = (float*)walloc((size_t)BATCH * NPTS * 4);
    int*   idx   = (int*)  walloc((size_t)BATCH * NPTS * KNN * 4);
    float* ymax  = (float*)walloc((size_t)BATCH * 256 * NPTS * 4);
    float* x1    = (float*)walloc((size_t)BATCH * 64 * NPTS * 4);
    float* x2    = (float*)walloc((size_t)BATCH * 64 * NPTS * 4);
    float* x3    = (float*)walloc((size_t)BATCH * 128 * NPTS * 4);
    float* x4    = (float*)walloc((size_t)BATCH * 256 * NPTS * 4);
    __hip_bfloat16* Wbf = (__hip_bfloat16*)walloc((size_t)1024 * 512 * 2);
    float* stats = (float*)walloc(2 * 1024 * 4);
    float* mean5 = (float*)walloc(1024 * 4);
    float* rstd5 = (float*)walloc(1024 * 4);
    float* pbuf  = (float*)walloc((size_t)BATCH * 2048 * 4);
    float* zbuf  = (float*)walloc((size_t)BATCH * 256 * 4);
    __hip_bfloat16* Fb = (__hip_bfloat16*)walloc((size_t)LB * 512 * 2);
    float* Yb    = (float*)walloc((size_t)256 * LB * 4);
    __hip_bfloat16* hbf = (__hip_bfloat16*)walloc((size_t)BATCH * NPTS * 512 * 2);
    float* Y5    = (float*)walloc((size_t)1024 * BATCH * NPTS * 4);

    struct Blk { const float* xin; int cin; int C2p; int O;
                 const float* W; const float* g; const float* be; float* xout; };
    Blk blks[4] = {
        { x,  3,   32,  64,  W1, g1, b1, x1 },
        { x1, 64,  128, 64,  W2, g2, b2, x2 },
        { x2, 64,  128, 128, W3, g3, b3, x3 },
        { x3, 128, 256, 256, W4, g4, b4, x4 },
    };

    for (int blk = 0; blk < 4; ++blk) {
        const Blk& Bk = blks[blk];
        int C = Bk.cin, O = Bk.O, C2p = Bk.C2p;
        xx_kernel<<<(BATCH * NPTS) / 256, 256, 0, stream>>>(Bk.xin, xx, C);
        size_t smem = ((size_t)256 * (C + 1) + (size_t)C * 128) * 4;
        knn_kernel<<<dim3(NPTS / 256, BATCH), 256, smem, stream>>>(Bk.xin, xx, idx, C);
        wcvt_kernel<<<(O * C2p + 255) / 256, 256, 0, stream>>>(Bk.W, Wbf, O, 2 * C, C2p);
        zero_kernel<<<(2048 + 255) / 256, 256, 0, stream>>>(stats, 2048);
        for (int b = 0; b < BATCH; ++b) {
            feat_kernel<<<LB / 256, 256, 0, stream>>>(Bk.xin, idx, Fb, b, C, C2p);
            wmma_gemm_kernel<<<dim3(LB / 128, O / 64), 256, 0, stream>>>(Wbf, Fb, Yb, O, C2p, LB);
            epilogue_kernel<<<dim3(NPTS / 256, O), 256, 0, stream>>>(Yb, ymax, stats, b, O);
        }
        apply_bn_kernel<<<(unsigned)(((size_t)BATCH * O * NPTS) / 256), 256, 0, stream>>>(
            ymax, stats, Bk.g, Bk.be, Bk.xout, O);
    }

    pack_h_kernel<<<(BATCH * NPTS) / 256, 256, 0, stream>>>(x1, x2, x3, x4, hbf);
    wcvt_kernel<<<(1024 * 512 + 255) / 256, 256, 0, stream>>>(W5, Wbf, 1024, 512, 512);
    wmma_gemm_kernel<<<dim3((BATCH * NPTS) / 128, 1024 / 64), 256, 0, stream>>>(
        Wbf, hbf, Y5, 1024, 512, BATCH * NPTS);
    stats5_kernel<<<1024, 256, 0, stream>>>(Y5, mean5, rstd5);
    pool_kernel<<<(BATCH * 1024) / 256, 256, 0, stream>>>(Y5, mean5, rstd5, g5, b5, pbuf);
    fc_kernel<<<(BATCH * 256) / 256, 256, 0, stream>>>(pbuf, Wl, zbuf);
    final_bn_kernel<<<1, 256, 0, stream>>>(zbuf, g6, b6, out);
}
